// GAT_Net_18056042512584
// MI455X (gfx1250) — matrix operand order repaired and checked by reading, not compiled, not run on hardware
//
#include <hip/hip_runtime.h>
#include <hip/hip_bf16.h>
#include <cstddef>

typedef __attribute__((ext_vector_type(16))) __bf16 v16bf;
typedef __attribute__((ext_vector_type(8)))  float  v8f;

// Problem constants (match reference)
constexpr int   Nn    = 50000;
constexpr int   Ee    = 1600000;
constexpr int   Edec  = 500000;
constexpr int   Etot  = Ee + Nn;        // edges + self loops
constexpr float SLOPE = 0.2f;
constexpr int   LDSSTR = 132;           // padded LDS row stride (floats): 132%64=4 -> conflict-free

// ---------- helpers ----------
__device__ __forceinline__ unsigned f2ord(float f) {
    unsigned u = __float_as_uint(f);
    return (u & 0x80000000u) ? ~u : (u | 0x80000000u);
}
__device__ __forceinline__ float ord2f(unsigned u) {
    return __uint_as_float((u & 0x80000000u) ? (u ^ 0x80000000u) : ~u);
}

__global__ void fill_zero(float* p, size_t n) {
    size_t i = (size_t)blockIdx.x * blockDim.x + threadIdx.x;
    if (i < n) p[i] = 0.0f;
}

// ---------- pack W[KDIM,NOUT] (f32, row-major) into bf16 B-fragments ----------
// Fragment layout: frag[(c*(NOUT/16) + nt)*32 + lane] = v16bf with
//   element e = W[(c*32 + half*8 + e + (e>=8 ? 8 : 0))*NOUT + nt*16 + (lane&15)],  half = lane>>4
// i.e. exactly the per-lane register image the WMMA B operand needs.
template<int KDIM, int NOUT>
__global__ void pack_wB(const float* __restrict__ W, v16bf* __restrict__ out) {
    constexpr int TOT = (KDIM / 32) * (NOUT / 16) * 32;
    int t = blockIdx.x * blockDim.x + threadIdx.x;
    if (t >= TOT) return;
    int lane = t & 31;
    int rest = t >> 5;
    int nt = rest % (NOUT / 16);
    int c  = rest / (NOUT / 16);
    int n = lane & 15, half = lane >> 4;
    int k0 = c * 32, n0 = nt * 16;
    v16bf b;
#pragma unroll
    for (int e = 0; e < 16; ++e)
        b[e] = (__bf16)W[(size_t)(k0 + half * 8 + e + (e >= 8 ? 8 : 0)) * NOUT + n0 + n];
    out[t] = b;
}

// ---------- node linear transform: C[M,NOUT] = X[M,KDIM] @ W[KDIM,NOUT] ----------
// grid.x = M/16 row tiles; blockDim.x = (NOUT/16)*32, one wave per 16-col tile.
// Wp = pre-packed bf16 B-fragments.
template<int KDIM, int NOUT>
__global__ __launch_bounds__((NOUT / 16) * 32)
void gemm_nodes(const float* __restrict__ X, const v16bf* __restrict__ Wp,
                float* __restrict__ C) {
    const int lane = threadIdx.x & 31;
    const int wave = threadIdx.x >> 5;
    const int m    = lane & 15;
    const int half = lane >> 4;
    const int n0   = wave * 16;
    const size_t row = (size_t)blockIdx.x * 16 + m;

    v8f acc = {};
    for (int k0 = 0; k0 < KDIM; k0 += 32) {
        v16bf a;
        const float4* arow =
            reinterpret_cast<const float4*>(X + row * KDIM + k0 + half * 8);
        float4 f0 = arow[0], f1 = arow[1];   // K = half*8 + 0..7
        float4 f2 = arow[4], f3 = arow[5];   // K = 16 + half*8 + 0..7
        a[0] = (__bf16)f0.x; a[1] = (__bf16)f0.y; a[2] = (__bf16)f0.z; a[3] = (__bf16)f0.w;
        a[4] = (__bf16)f1.x; a[5] = (__bf16)f1.y; a[6] = (__bf16)f1.z; a[7] = (__bf16)f1.w;
        a[8]  = (__bf16)f2.x; a[9]  = (__bf16)f2.y; a[10] = (__bf16)f2.z; a[11] = (__bf16)f2.w;
        a[12] = (__bf16)f3.x; a[13] = (__bf16)f3.y; a[14] = (__bf16)f3.z; a[15] = (__bf16)f3.w;
        v16bf b = Wp[(size_t)((k0 >> 5) * (NOUT / 16) + wave) * 32 + lane];
        acc = __builtin_amdgcn_wmma_f32_16x16x32_bf16(false, a, false, b,
                                                      (short)0, acc, false, false);
    }
#pragma unroll
    for (int v = 0; v < 8; ++v) {
        size_t r = (size_t)blockIdx.x * 16 + v + half * 8;
        C[r * NOUT + n0 + m] = acc[v];
    }
}

// ---------- edge scores: e = leaky_relu(xl[src]+xr[dst]) . att  (8 lanes/edge) ----------
template<int F>
__global__ void edge_scores(const float* __restrict__ xl, const float* __restrict__ xr,
                            const float* __restrict__ att, const int* __restrict__ ei,
                            float* __restrict__ ebuf) {
    int t = blockIdx.x * blockDim.x + threadIdx.x;
    int edge = t >> 3, sub = t & 7;
    if (edge >= Etot) return;
    int s, d;
    if (edge < Ee) { s = ei[edge]; d = ei[Ee + edge]; }
    else           { s = d = edge - Ee; }
    constexpr int CH = F / 8;
    const float* pl = xl + (size_t)s * F + sub * CH;
    const float* pr = xr + (size_t)d * F + sub * CH;
    const float* pa = att + sub * CH;
    float p = 0.0f;
#pragma unroll
    for (int j = 0; j < CH; ++j) {
        float v = pl[j] + pr[j];
        v = v > 0.0f ? v : SLOPE * v;
        p += v * pa[j];
    }
#pragma unroll
    for (int o = 4; o; o >>= 1) p += __shfl_xor(p, o, 8);
    if (sub == 0) ebuf[edge] = p;
}

// ---------- segment max over dst ----------
__global__ void seg_max(const float* __restrict__ ebuf, const int* __restrict__ ei,
                        unsigned* __restrict__ mbuf) {
    int edge = blockIdx.x * blockDim.x + threadIdx.x;
    if (edge >= Etot) return;
    int d = (edge < Ee) ? ei[Ee + edge] : edge - Ee;
    atomicMax(&mbuf[d], f2ord(ebuf[edge]));
}

// ---------- w = exp(e - m[dst]);  denom[dst] += w ----------
__global__ void seg_expsum(float* __restrict__ ebuf, const int* __restrict__ ei,
                           const unsigned* __restrict__ mbuf, float* __restrict__ denom) {
    int edge = blockIdx.x * blockDim.x + threadIdx.x;
    if (edge >= Etot) return;
    int d = (edge < Ee) ? ei[Ee + edge] : edge - Ee;
    float w = __expf(ebuf[edge] - ord2f(mbuf[d]));
    ebuf[edge] = w;
    atomicAdd(&denom[d], w);
}

// ---------- out[dst] += (w/denom[dst]) * xl[src]  (8 lanes/edge) ----------
template<int F>
__global__ void seg_scatter(const float* __restrict__ ebuf, const float* __restrict__ denom,
                            const float* __restrict__ xl, const int* __restrict__ ei,
                            float* __restrict__ out) {
    int t = blockIdx.x * blockDim.x + threadIdx.x;
    int edge = t >> 3, sub = t & 7;
    if (edge >= Etot) return;
    int s, d;
    if (edge < Ee) { s = ei[edge]; d = ei[Ee + edge]; }
    else           { s = d = edge - Ee; }
    float alpha = ebuf[edge] / denom[d];
    constexpr int CH = F / 8;
    const float* pl = xl + (size_t)s * F + sub * CH;
    float*       po = out + (size_t)d * F + sub * CH;
#pragma unroll
    for (int j = 0; j < CH; ++j) atomicAdd(&po[j], alpha * pl[j]);
}

// ---------- bias (+optional relu) ----------
template<int F, bool RELU>
__global__ void bias_act(float* __restrict__ z, const float* __restrict__ bias, size_t n) {
    size_t i = (size_t)blockIdx.x * blockDim.x + threadIdx.x;
    if (i >= n) return;
    float v = z[i] + bias[i & (size_t)(F - 1)];
    if (RELU) v = fmaxf(v, 0.0f);
    z[i] = v;
}

// ---------- fused decoder: tile of 16 rows through 4 MLP layers ----------
// A: LDS [16][K] (stride LDSSTR, f32) ; Wp: packed bf16 B-fragments ; Out: LDS [16][NOUT]
template<int K, int NOUT>
__device__ __forceinline__ void tile_gemm(const float* __restrict__ A,
                                          const v16bf* __restrict__ Wp,
                                          const float* __restrict__ bias,
                                          float* __restrict__ Out) {
    const int lane = threadIdx.x & 31;
    const int m = lane & 15, half = lane >> 4;
#pragma unroll
    for (int nt = 0; nt < NOUT / 16; ++nt) {
        const int n0 = nt * 16;
        v8f acc = {};
#pragma unroll
        for (int c = 0; c < K / 32; ++c) {
            const int k0 = c * 32;
            v16bf a;
            const float* arow = A + m * LDSSTR + k0 + half * 8;
#pragma unroll
            for (int j = 0; j < 8; ++j) {
                a[j]     = (__bf16)arow[j];
                a[8 + j] = (__bf16)arow[16 + j];
            }
            v16bf b = Wp[(size_t)(c * (NOUT / 16) + nt) * 32 + lane];
            acc = __builtin_amdgcn_wmma_f32_16x16x32_bf16(false, a, false, b,
                                                          (short)0, acc, false, false);
        }
#pragma unroll
        for (int v = 0; v < 8; ++v) {
            float val = fmaxf(acc[v] + bias[n0 + m], 0.0f);  // relu on layers 1-3
            Out[(v + half * 8) * LDSSTR + n0 + m] = val;
        }
    }
}

__global__ __launch_bounds__(32)
void decoder_kernel(const float* __restrict__ z, const int* __restrict__ pos,
                    const int* __restrict__ neg,
                    const v16bf* __restrict__ fw1p, const float* __restrict__ fb1,
                    const v16bf* __restrict__ fw2p, const float* __restrict__ fb2,
                    const v16bf* __restrict__ fw3p, const float* __restrict__ fb3,
                    const float* __restrict__ fw4, const float* __restrict__ fb4,
                    float* __restrict__ out) {
    __shared__ float bufA[16 * LDSSTR];
    __shared__ float bufB[16 * LDSSTR];
    const int tile = blockIdx.x;
    const int lane = threadIdx.x;

    // stage concat(z[a], z[b]) for 16 rows into LDS
    for (int r = 0; r < 16; ++r) {
        int eid = tile * 16 + r;
        int ia, ib;
        if (eid < Edec) { ia = pos[eid];        ib = pos[Edec + eid]; }
        else            { ia = neg[eid - Edec]; ib = neg[Edec + eid - Edec]; }
        for (int f = lane; f < 128; f += 32)
            bufA[r * LDSSTR + f] = (f < 64) ? z[(size_t)ia * 64 + f]
                                            : z[(size_t)ib * 64 + f - 64];
    }
    __syncthreads();
    tile_gemm<128, 64>(bufA, fw1p, fb1, bufB);   // 128 -> 64, relu
    __syncthreads();
    tile_gemm<64, 128>(bufB, fw2p, fb2, bufA);   // 64 -> 128, relu
    __syncthreads();
    tile_gemm<128, 64>(bufA, fw3p, fb3, bufB);   // 128 -> 64, relu
    __syncthreads();
    if (lane < 16) {                             // 64 -> 1
        float s = fb4[0];
#pragma unroll 8
        for (int f = 0; f < 64; ++f) s += bufB[lane * LDSSTR + f] * fw4[f];
        out[(size_t)tile * 16 + lane] = s;
    }
}

// ---------------------------------------------------------------------------
extern "C" void kernel_launch(void* const* d_in, const int* in_sizes, int n_in,
                              void* d_out, int out_size, void* d_ws, size_t ws_size,
                              hipStream_t stream) {
    (void)in_sizes; (void)n_in; (void)out_size; (void)ws_size;
    const float* x    = (const float*)d_in[0];
    const int*   ei   = (const int*)  d_in[1];
    const int*   pos  = (const int*)  d_in[2];
    const int*   neg  = (const int*)  d_in[3];
    const float* w1l  = (const float*)d_in[4];
    const float* w1r  = (const float*)d_in[5];
    const float* a1   = (const float*)d_in[6];
    const float* b1   = (const float*)d_in[7];
    const float* w2l  = (const float*)d_in[8];
    const float* w2r  = (const float*)d_in[9];
    const float* a2   = (const float*)d_in[10];
    const float* b2   = (const float*)d_in[11];
    const float* fw1  = (const float*)d_in[12];
    const float* fb1  = (const float*)d_in[13];
    const float* fw2  = (const float*)d_in[14];
    const float* fb2  = (const float*)d_in[15];
    const float* fw3  = (const float*)d_in[16];
    const float* fb3  = (const float*)d_in[17];
    const float* fw4  = (const float*)d_in[18];
    const float* fb4  = (const float*)d_in[19];
    float* out = (float*)d_out;

    // workspace carve-out
    char* ws = (char*)d_ws;
    size_t off = 0;
    auto carve = [&](size_t bytes) -> void* {
        void* p = ws + off;
        off += (bytes + 255) & ~(size_t)255;
        return p;
    };
    float*    xl1   = (float*)   carve((size_t)Nn * 128 * 4);
    float*    xr1   = (float*)   carve((size_t)Nn * 128 * 4);
    float*    z1    = (float*)   carve((size_t)Nn * 128 * 4);
    float*    xl2   = (float*)   carve((size_t)Nn * 64 * 4);
    float*    xr2   = (float*)   carve((size_t)Nn * 64 * 4);
    float*    z2    = (float*)   carve((size_t)Nn * 64 * 4);
    float*    ebuf  = (float*)   carve((size_t)Etot * 4);
    unsigned* mbuf  = (unsigned*)carve((size_t)Nn * 4);
    float*    denom = (float*)   carve((size_t)Nn * 4);
    // packed bf16 B-fragments (32 B per fragment-lane)
    v16bf* w1lp = (v16bf*)carve((size_t)4 * 8 * 32 * 32);  // 128x128
    v16bf* w1rp = (v16bf*)carve((size_t)4 * 8 * 32 * 32);
    v16bf* w2lp = (v16bf*)carve((size_t)4 * 4 * 32 * 32);  // 128x64
    v16bf* w2rp = (v16bf*)carve((size_t)4 * 4 * 32 * 32);
    v16bf* fw1p = (v16bf*)carve((size_t)4 * 4 * 32 * 32);  // 128x64
    v16bf* fw2p = (v16bf*)carve((size_t)2 * 8 * 32 * 32);  // 64x128
    v16bf* fw3p = (v16bf*)carve((size_t)4 * 4 * 32 * 32);  // 128x64

    auto blocks = [](size_t n, int b) { return (unsigned)((n + b - 1) / b); };
    const int eth8 = 8 * Etot;   // 8 lanes per edge

    // ---- pack weights into bf16 fragment layout (tiny) ----
    pack_wB<128, 128><<<blocks(4 * 8 * 32, 256), 256, 0, stream>>>(w1l, w1lp);
    pack_wB<128, 128><<<blocks(4 * 8 * 32, 256), 256, 0, stream>>>(w1r, w1rp);
    pack_wB<128, 64><<<blocks(4 * 4 * 32, 256), 256, 0, stream>>>(w2l, w2lp);
    pack_wB<128, 64><<<blocks(4 * 4 * 32, 256), 256, 0, stream>>>(w2r, w2rp);
    pack_wB<128, 64><<<blocks(4 * 4 * 32, 256), 256, 0, stream>>>(fw1, fw1p);
    pack_wB<64, 128><<<blocks(2 * 8 * 32, 256), 256, 0, stream>>>(fw2, fw2p);
    pack_wB<128, 64><<<blocks(4 * 4 * 32, 256), 256, 0, stream>>>(fw3, fw3p);

    // ---- layer 1 (128 -> 128) ----
    gemm_nodes<128, 128><<<Nn / 16, 256, 0, stream>>>(x, w1lp, xl1);
    gemm_nodes<128, 128><<<Nn / 16, 256, 0, stream>>>(x, w1rp, xr1);
    fill_zero<<<blocks((size_t)Nn * 128, 256), 256, 0, stream>>>(z1, (size_t)Nn * 128);
    fill_zero<<<blocks(Nn, 256), 256, 0, stream>>>((float*)mbuf, Nn);
    fill_zero<<<blocks(Nn, 256), 256, 0, stream>>>(denom, Nn);
    edge_scores<128><<<blocks(eth8, 256), 256, 0, stream>>>(xl1, xr1, a1, ei, ebuf);
    seg_max   <<<blocks(Etot, 256), 256, 0, stream>>>(ebuf, ei, mbuf);
    seg_expsum<<<blocks(Etot, 256), 256, 0, stream>>>(ebuf, ei, mbuf, denom);
    seg_scatter<128><<<blocks(eth8, 256), 256, 0, stream>>>(ebuf, denom, xl1, ei, z1);
    bias_act<128, true><<<blocks((size_t)Nn * 128, 256), 256, 0, stream>>>(z1, b1, (size_t)Nn * 128);

    // ---- layer 2 (128 -> 64) ----
    gemm_nodes<128, 64><<<Nn / 16, 128, 0, stream>>>(z1, w2lp, xl2);
    gemm_nodes<128, 64><<<Nn / 16, 128, 0, stream>>>(z1, w2rp, xr2);
    fill_zero<<<blocks((size_t)Nn * 64, 256), 256, 0, stream>>>(z2, (size_t)Nn * 64);
    fill_zero<<<blocks(Nn, 256), 256, 0, stream>>>((float*)mbuf, Nn);
    fill_zero<<<blocks(Nn, 256), 256, 0, stream>>>(denom, Nn);
    edge_scores<64><<<blocks(eth8, 256), 256, 0, stream>>>(xl2, xr2, a2, ei, ebuf);
    seg_max   <<<blocks(Etot, 256), 256, 0, stream>>>(ebuf, ei, mbuf);
    seg_expsum<<<blocks(Etot, 256), 256, 0, stream>>>(ebuf, ei, mbuf, denom);
    seg_scatter<64><<<blocks(eth8, 256), 256, 0, stream>>>(ebuf, denom, xl2, ei, z2);
    bias_act<64, false><<<blocks((size_t)Nn * 64, 256), 256, 0, stream>>>(z2, b2, (size_t)Nn * 64);

    // ---- fused decoder MLP over 1M edge pairs ----
    decoder_kernel<<<(2 * Edec) / 16, 32, 0, stream>>>(z2, pos, neg,
                                                       fw1p, fb1, fw2p, fb2,
                                                       fw3p, fb3, fw4, fb4, out);
}
